// GateLoop_60387240182219
// MI455X (gfx1250) — matrix operand order repaired and checked by reading, not compiled, not run on hardware
//
#include <hip/hip_runtime.h>

// ---------------------------------------------------------------------------
// GateLoop for MI455X (gfx1250, wave32, WMMA + async-to-LDS double buffering).
// Pipeline:
//   1) rmsnorm_cast:    x -> xn (bf16)
//   2) transpose_cast:  w* (KxN f32) -> wt* (NxK bf16)
//   3) wmma_gemm x5:    q,k,v, a=sigmoid, gs=silu(g)  (bf16 x bf16 -> f32)
//   4) mul:             kv = k*v
//   5) chunked scan (3 passes), fused with q*y*silu(g), emits yp (bf16)
//   6) wmma_gemm:       out = yp @ wo  (f32 out)
// GEMM inner loop: LDS-staged tiles filled with global_load_async_to_lds_b128
// (ASYNCcnt), double buffered, consumed by v_wmma_f32_16x16x32_bf16.
// ---------------------------------------------------------------------------

typedef __attribute__((ext_vector_type(16))) __bf16 v16bf;
typedef __attribute__((ext_vector_type(8)))  float  v8f;

#define B_SZ    2
#define S_LEN   4096
#define D_DIM   1024
#define M_ROWS  (B_SZ * S_LEN)     // 8192
#define CHUNK   32
#define NCHUNK  (S_LEN / CHUNK)    // 128

// LDS tile geometry (elements are bf16 = 2B)
#define KT        32               // K per stage
#define LDS_PITCH 40               // 32 + 8 pad -> 80B row stride, conflict-floor b128 reads
#define A_TILE_E  (256 * LDS_PITCH)   // 10240 elems = 20480 B
#define B_TILE_E  (128 * LDS_PITCH)   //  5120 elems = 10240 B

__device__ __forceinline__ unsigned short f2bf(float f) {
  unsigned int u = __float_as_uint(f);
  u += 0x7FFFu + ((u >> 16) & 1u);        // round-to-nearest-even
  return (unsigned short)(u >> 16);
}

// Low 32 bits of a flat LDS address are the LDS byte offset (ISA 10.2).
__device__ __forceinline__ unsigned lds_off(const void* p) {
  return (unsigned)(size_t)p;
}

__device__ __forceinline__ void async_b128_to_lds(unsigned lds, const void* g) {
  asm volatile("global_load_async_to_lds_b128 %0, %1, off"
               :: "v"(lds), "v"(g) : "memory");
}

// ------------------------- RMSNorm + bf16 cast -----------------------------
__global__ void rmsnorm_cast_kernel(const float* __restrict__ x,
                                    const float* __restrict__ gamma,
                                    unsigned short* __restrict__ xn) {
  const int row = blockIdx.x;
  const int tid = threadIdx.x;
  const float4 x4 = reinterpret_cast<const float4*>(x)[(size_t)row * (D_DIM / 4) + tid];
  float ss = x4.x * x4.x + x4.y * x4.y + x4.z * x4.z + x4.w * x4.w;
#pragma unroll
  for (int off = 16; off > 0; off >>= 1) ss += __shfl_xor(ss, off, 32);
  __shared__ float red[8];
  const int wid = tid >> 5, lane = tid & 31;
  if (lane == 0) red[wid] = ss;
  __syncthreads();
  const float tot = red[0] + red[1] + red[2] + red[3] +
                    red[4] + red[5] + red[6] + red[7];
  const float inv = rsqrtf(tot + 1e-5f) * 32.0f;    // rsqrt(sum_sq+eps)*sqrt(1024)
  const float4 g4 = reinterpret_cast<const float4*>(gamma)[tid];
  ushort4 o;
  o.x = f2bf(x4.x * inv * g4.x);
  o.y = f2bf(x4.y * inv * g4.y);
  o.z = f2bf(x4.z * inv * g4.z);
  o.w = f2bf(x4.w * inv * g4.w);
  reinterpret_cast<ushort4*>(xn)[(size_t)row * (D_DIM / 4) + tid] = o;
}

// -------------------- weight transpose + bf16 cast -------------------------
__global__ void transpose_cast_kernel(const float* __restrict__ w,
                                      unsigned short* __restrict__ wt) {
  __shared__ float tile[32][33];
  const int tx = threadIdx.x, ty = threadIdx.y;   // 32 x 8
  const int n0 = blockIdx.x * 32, k0 = blockIdx.y * 32;
#pragma unroll
  for (int j = 0; j < 32; j += 8)
    tile[ty + j][tx] = w[(size_t)(k0 + ty + j) * D_DIM + n0 + tx];
  __syncthreads();
#pragma unroll
  for (int j = 0; j < 32; j += 8)
    wt[(size_t)(n0 + ty + j) * D_DIM + k0 + tx] = f2bf(tile[tx][ty + j]);
}

// ------------------------------ WMMA GEMM ----------------------------------
// C(MxN,f32) = act( A(MxK,bf16) @ Bt(NxK,bf16)^T )
// 256 thr = 8 waves (4 M x 2 N); wave tile 64x64 = 4x4 wmma frags.
// A/B tiles staged in LDS via async copies, double buffered.
union Frag16 { v16bf v; uint4 u[2]; };

__device__ __forceinline__ void issue_tile_copies(
    const unsigned short* __restrict__ A, const unsigned short* __restrict__ Bt,
    int K, int m0blk, int n0blk, int k0,
    unsigned short* aBuf, unsigned short* bBuf, int tid) {
  // A tile: 256 rows x 32 K-elems = 1024 16B chunks; 4 per thread.
#pragma unroll
  for (int i = 0; i < 4; ++i) {
    const int id = i * 256 + tid;
    const int row = id >> 2, c = id & 3;          // 4 chunks per row
    const unsigned short* g = A + (size_t)(m0blk + row) * K + k0 + c * 8;
    async_b128_to_lds(lds_off(aBuf + row * LDS_PITCH + c * 8), g);
  }
  // B tile: 128 rows x 32 K-elems = 512 16B chunks; 2 per thread.
#pragma unroll
  for (int i = 0; i < 2; ++i) {
    const int id = i * 256 + tid;
    const int row = id >> 2, c = id & 3;
    const unsigned short* g = Bt + (size_t)(n0blk + row) * K + k0 + c * 8;
    async_b128_to_lds(lds_off(bBuf + row * LDS_PITCH + c * 8), g);
  }
}

__global__ void __launch_bounds__(256, 1)
wmma_gemm_kernel(const unsigned short* __restrict__ A,
                 const unsigned short* __restrict__ Bt,
                 float* __restrict__ C,
                 int M, int N, int K, int act) {
  __shared__ __align__(16) unsigned short smem[2 * A_TILE_E + 2 * B_TILE_E];

  const int tid  = threadIdx.x;
  const int lane = tid & 31;
  const int wave = tid >> 5;
  const int wm = wave & 3;                   // 4 waves along M
  const int wn = wave >> 2;                  // 2 waves along N
  const int m0blk = blockIdx.y * 256;
  const int n0blk = blockIdx.x * 128;
  const int half = lane >> 4;
  const int l15  = lane & 15;

  const v8f zero = {0.f, 0.f, 0.f, 0.f, 0.f, 0.f, 0.f, 0.f};
  v8f acc[4][4];
#pragma unroll
  for (int i = 0; i < 4; ++i)
#pragma unroll
    for (int j = 0; j < 4; ++j) acc[i][j] = zero;

  // Prologue: fill buffer 0. (Buffer bases computed arithmetically each time —
  // no LDS-pointer arrays, which would force an illegal static initializer.)
  issue_tile_copies(A, Bt, K, m0blk, n0blk, 0,
                    smem, smem + 2 * A_TILE_E, tid);

  for (int k0 = 0; k0 < K; k0 += KT) {
    const int cur = (k0 / KT) & 1;
    unsigned short* aCur = smem + cur * A_TILE_E;
    unsigned short* bCur = smem + 2 * A_TILE_E + cur * B_TILE_E;

    if (k0 + KT < K) {
      unsigned short* aNxt = smem + (cur ^ 1) * A_TILE_E;
      unsigned short* bNxt = smem + 2 * A_TILE_E + (cur ^ 1) * B_TILE_E;
      issue_tile_copies(A, Bt, K, m0blk, n0blk, k0 + KT, aNxt, bNxt, tid);
      // 12 async copies in flight; first 6 (current buffer) retire in order.
      asm volatile("s_wait_asynccnt 0x6" ::: "memory");
    } else {
      asm volatile("s_wait_asynccnt 0x0" ::: "memory");
    }
    __syncthreads();                         // current tile visible WG-wide

    Frag16 af[4], bfr[4];
#pragma unroll
    for (int f = 0; f < 4; ++f) {
      // A frag: lane holds K = half*8+{0..7} and 16+half*8+{0..7}
      const unsigned short* ap =
          aCur + (wm * 64 + f * 16 + l15) * LDS_PITCH + half * 8;
      af[f].u[0] = *reinterpret_cast<const uint4*>(ap);
      af[f].u[1] = *reinterpret_cast<const uint4*>(ap + 16);
      // B frag: lane = column, K = half*8+{0..7} and 16+half*8+{0..7}
      const unsigned short* bp =
          bCur + (wn * 64 + f * 16 + l15) * LDS_PITCH + half * 8;
      bfr[f].u[0] = *reinterpret_cast<const uint4*>(bp);
      bfr[f].u[1] = *reinterpret_cast<const uint4*>(bp + 16);
    }
#pragma unroll
    for (int i = 0; i < 4; ++i)
#pragma unroll
      for (int j = 0; j < 4; ++j)
        acc[i][j] = __builtin_amdgcn_wmma_f32_16x16x32_bf16(
            false, af[i].v, false, bfr[j].v, (short)0, acc[i][j], false, false);

    __syncthreads();                         // done reading before next overwrite
  }

#pragma unroll
  for (int i = 0; i < 4; ++i) {
    const int mrow = m0blk + wm * 64 + i * 16 + half * 8;  // D: M = r + 8*half
#pragma unroll
    for (int j = 0; j < 4; ++j) {
      const int ncol = n0blk + wn * 64 + j * 16 + l15;
#pragma unroll
      for (int r = 0; r < 8; ++r) {
        float v = acc[i][j][r];
        if (act == 1)      v = 1.0f / (1.0f + __expf(-v));       // sigmoid
        else if (act == 2) v = v / (1.0f + __expf(-v));          // silu
        C[(size_t)(mrow + r) * N + ncol] = v;
      }
    }
  }
}

// ------------------------------ kv = k * v ---------------------------------
__global__ void mul_kernel(const float* __restrict__ k,
                           const float* __restrict__ v,
                           float* __restrict__ kv) {
  const size_t i = (size_t)blockIdx.x * 256 + threadIdx.x;
  const float4 a = reinterpret_cast<const float4*>(k)[i];
  const float4 b = reinterpret_cast<const float4*>(v)[i];
  float4 o;
  o.x = a.x * b.x; o.y = a.y * b.y; o.z = a.z * b.z; o.w = a.w * b.w;
  reinterpret_cast<float4*>(kv)[i] = o;
}

// ------------------------ chunked gated scan -------------------------------
__global__ void scan_pass1_kernel(const float* __restrict__ a,
                                  const float* __restrict__ kv,
                                  float* __restrict__ P,
                                  float* __restrict__ Ye) {
  const int d = blockIdx.x * 256 + threadIdx.x;
  const int c = blockIdx.y;
  const int b = blockIdx.z;
  const size_t base = ((size_t)b * S_LEN + (size_t)c * CHUNK) * D_DIM + d;
  float y = 0.f, p = 1.f;
#pragma unroll 4
  for (int i = 0; i < CHUNK; ++i) {
    const float av  = a[base + (size_t)i * D_DIM];
    const float kvv = kv[base + (size_t)i * D_DIM];
    y = fmaf(av, y, kvv);
    p *= av;
  }
  const size_t o = ((size_t)b * NCHUNK + c) * D_DIM + d;
  P[o] = p;
  Ye[o] = y;
}

__global__ void scan_pass2_kernel(const float* __restrict__ P,
                                  const float* __restrict__ Ye,
                                  float* __restrict__ Cy) {
  const int d = blockIdx.x * 256 + threadIdx.x;
  const int b = blockIdx.y;
  float carry = 0.f;
  for (int c = 0; c < NCHUNK; ++c) {
    const size_t o = ((size_t)b * NCHUNK + c) * D_DIM + d;
    Cy[o] = carry;
    carry = fmaf(P[o], carry, Ye[o]);
  }
}

__global__ void scan_pass3_kernel(const float* __restrict__ a,
                                  const float* __restrict__ kv,
                                  const float* __restrict__ q,
                                  const float* __restrict__ gs,
                                  const float* __restrict__ Cy,
                                  unsigned short* __restrict__ yp) {
  const int d = blockIdx.x * 256 + threadIdx.x;
  const int c = blockIdx.y;
  const int b = blockIdx.z;
  const size_t base = ((size_t)b * S_LEN + (size_t)c * CHUNK) * D_DIM + d;
  float y = Cy[((size_t)b * NCHUNK + c) * D_DIM + d];
#pragma unroll 4
  for (int i = 0; i < CHUNK; ++i) {
    const size_t idx = base + (size_t)i * D_DIM;
    y = fmaf(a[idx], y, kv[idx]);
    yp[idx] = f2bf(q[idx] * y * gs[idx]);   // q * y * silu(g)
  }
}

// ------------------------------- launch ------------------------------------
extern "C" void kernel_launch(void* const* d_in, const int* in_sizes, int n_in,
                              void* d_out, int out_size, void* d_ws, size_t ws_size,
                              hipStream_t stream) {
  (void)in_sizes; (void)n_in; (void)out_size; (void)ws_size;

  const float* x     = (const float*)d_in[0];
  const float* gamma = (const float*)d_in[1];
  // weights order: wq, wk, wv, wa, wg, wo  -> d_in[2..7]

  const size_t MD  = (size_t)M_ROWS * D_DIM;
  const size_t DD  = (size_t)D_DIM * D_DIM;
  const size_t BCD = (size_t)B_SZ * NCHUNK * D_DIM;

  char* w = (char*)d_ws;
  unsigned short* xn = (unsigned short*)w;  w += MD * 2;
  unsigned short* wt = (unsigned short*)w;  w += 6 * DD * 2;
  float* qb = (float*)w;  w += MD * 4;
  float* ab = (float*)w;  w += MD * 4;
  float* gs = (float*)w;  w += MD * 4;
  float* kb = (float*)w;  w += MD * 4;
  float* vb = (float*)w;  w += MD * 4;   // becomes kv after mul_kernel
  float* Pb = (float*)w;  w += BCD * 4;
  float* Ye = (float*)w;  w += BCD * 4;
  float* Cy = (float*)w;  w += BCD * 4;
  unsigned short* yp = (unsigned short*)w;  w += MD * 2;

  rmsnorm_cast_kernel<<<M_ROWS, 256, 0, stream>>>(x, gamma, xn);

  {
    dim3 tb(32, 8), tg(D_DIM / 32, D_DIM / 32);
    for (int i = 0; i < 6; ++i)
      transpose_cast_kernel<<<tg, tb, 0, stream>>>((const float*)d_in[2 + i],
                                                   wt + (size_t)i * DD);
  }

  dim3 gblk(256), ggrd(D_DIM / 128, M_ROWS / 256);
  wmma_gemm_kernel<<<ggrd, gblk, 0, stream>>>(xn, wt + 0 * DD, qb, M_ROWS, D_DIM, D_DIM, 0);
  wmma_gemm_kernel<<<ggrd, gblk, 0, stream>>>(xn, wt + 1 * DD, kb, M_ROWS, D_DIM, D_DIM, 0);
  wmma_gemm_kernel<<<ggrd, gblk, 0, stream>>>(xn, wt + 2 * DD, vb, M_ROWS, D_DIM, D_DIM, 0);
  wmma_gemm_kernel<<<ggrd, gblk, 0, stream>>>(xn, wt + 3 * DD, ab, M_ROWS, D_DIM, D_DIM, 1);
  wmma_gemm_kernel<<<ggrd, gblk, 0, stream>>>(xn, wt + 4 * DD, gs, M_ROWS, D_DIM, D_DIM, 2);

  mul_kernel<<<(unsigned)(MD / 4 / 256), 256, 0, stream>>>(kb, vb, vb);

  scan_pass1_kernel<<<dim3(D_DIM / 256, NCHUNK, B_SZ), 256, 0, stream>>>(ab, vb, Pb, Ye);
  scan_pass2_kernel<<<dim3(D_DIM / 256, B_SZ), 256, 0, stream>>>(Pb, Ye, Cy);
  scan_pass3_kernel<<<dim3(D_DIM / 256, NCHUNK, B_SZ), 256, 0, stream>>>(ab, vb, qb, gs, Cy, yp);

  wmma_gemm_kernel<<<ggrd, gblk, 0, stream>>>(yp, wt + 5 * DD, (float*)d_out,
                                              M_ROWS, D_DIM, D_DIM, 0);
}
